// GeneralizedCollageOperator_4174708212007
// MI455X (gfx1250) — matrix operand order repaired and checked by reading, not compiled, not run on hardware
//
#include <hip/hip_runtime.h>
#include <hip/hip_bf16.h>
#include <cstdint>
#include <cstddef>

// ---------------- static problem config ----------------
#define BATCH   64
#define RES     32
#define CIN     3
#define MID     504      // hidden channels
#define OUTD    1008     // hypernet head outputs
#define CPAD    512      // padded channel count for WMMA (N and K padding)
#define XPAD    32       // padded input channels for conv1
#define NPIX    (RES*RES)

typedef __attribute__((ext_vector_type(16))) _Float16 v16h;
typedef __attribute__((ext_vector_type(8)))  float    v8f;
typedef __attribute__((ext_vector_type(4)))  float    f32x4;

union FragH { v16h h; f32x4 f[2]; };

// async global->LDS: prefer the clang builtin if this toolchain has it,
// otherwise drive GLOBAL_LOAD_ASYNC_TO_LDS_B128 / s_wait_asynccnt directly
// with inline asm (ISA 15.18.3 op 98; ASYNCcnt per ch. 15.5).
#if defined(__has_builtin)
#if __has_builtin(__builtin_amdgcn_load_to_lds) && __has_builtin(__builtin_amdgcn_s_wait_asynccnt)
#define USE_ASYNC_BUILTIN 1
#endif
#endif
#ifndef USE_ASYNC_BUILTIN
#define USE_ASYNC_BUILTIN 0
#endif

__device__ __forceinline__ float gelu_tanh(float x) {
    // jax.nn.gelu default (approximate=True)
    const float k0 = 0.7978845608028654f;
    float x3 = x * x * x;
    return 0.5f * x * (1.0f + tanhf(k0 * (x + 0.044715f * x3)));
}

// ---------------- packing kernels ----------------

// gelu(x) -> f16, channels padded 3 -> 32 (zeros)
__global__ __launch_bounds__(256) void pack_xg_kernel(const float* __restrict__ x,
                                                      _Float16* __restrict__ xg) {
    int idx = blockIdx.x * blockDim.x + threadIdx.x;     // BATCH*NPIX*XPAD
    if (idx >= BATCH * NPIX * XPAD) return;
    int c = idx & (XPAD - 1);
    int p = idx >> 5;                                    // b*NPIX + pixel
    float v = 0.0f;
    if (c < CIN) v = gelu_tanh(x[p * CIN + c]);
    xg[idx] = (_Float16)v;
}

// w1 (1,1,3,504) f32 -> B-fragment layout, single K-chunk of 32, N padded to 512.
// Fragment flat layout: [nb][lane][16 halves]; lane<16: K=v*2+h, lane>=16: +16; n = nb*16 + lane%16.
__global__ __launch_bounds__(256) void pack_w1_kernel(const float* __restrict__ w1,
                                                      _Float16* __restrict__ w1p) {
    int idx = blockIdx.x * blockDim.x + threadIdx.x;     // 32*512
    if (idx >= 32 * 512) return;
    int t    = idx & 15;
    int lane = (idx >> 4) & 31;
    int nb   = idx >> 9;
    int v = t >> 1, hh = t & 1;
    int k = v * 2 + hh + ((lane >= 16) ? 16 : 0);
    int n = nb * 16 + (lane & 15);
    float val = (k < CIN && n < MID) ? w1[k * MID + n] : 0.0f;
    w1p[idx] = (_Float16)val;
}

// w2/w3 (3,3,504,504) f32 -> [tap][kc][nb][lane][16 halves], K,N padded to 512
__global__ __launch_bounds__(256) void pack_w9_kernel(const float* __restrict__ w,
                                                      _Float16* __restrict__ wp) {
    int idx = blockIdx.x * blockDim.x + threadIdx.x;     // 9*16*32*512 = 2359296
    if (idx >= 9 * 16 * 32 * 512) return;
    int t    = idx & 15;
    int lane = (idx >> 4) & 31;
    int nb   = (idx >> 9) & 31;
    int kc   = (idx >> 14) & 15;
    int tap  = idx >> 18;
    int v = t >> 1, hh = t & 1;
    int kl = v * 2 + hh + ((lane >= 16) ? 16 : 0);
    int k = kc * 32 + kl;
    int n = nb * 16 + (lane & 15);
    float val = (k < MID && n < MID) ? w[((size_t)tap * MID + k) * MID + n] : 0.0f;
    wp[idx] = (_Float16)val;
}

// ---------------- WMMA implicit-GEMM conv ----------------
// One workgroup (8 waves) per (batch, image row). Input rows (with 1-pixel
// zero halo) staged in LDS via async LDS-DMA; wave w owns channels
// [w*64, w*64+64): 2 M-tiles (32 pixels) x 4 N-blocks, so B fragments in
// registers are reused across both M-tiles. Output restaged through LDS for
// coalesced b128 stores.
template<int TAPS, int KC, int CINP, bool GACT>
__global__ __launch_bounds__(256) void conv_wmma_kernel(const _Float16* __restrict__ ain,
                                                        const _Float16* __restrict__ wf,
                                                        const float*    __restrict__ bias,
                                                        _Float16*       __restrict__ aout) {
    constexpr int NR    = (TAPS == 9) ? 3 : 1;   // input rows incl. halo
    constexpr int R0    = (TAPS == 9) ? 1 : 0;
    constexpr int PITCH = CINP + 8;              // halves; 260 DWORDs -> conflict-free
    constexpr int SM_IN = NR * 34 * PITCH;
    constexpr int SM_OUT = 32 * CPAD;
    constexpr int SMSZ  = (SM_IN > SM_OUT) ? SM_IN : SM_OUT;
    __shared__ _Float16 smem[SMSZ];

    const int row  = blockIdx.x;                 // 0 .. BATCH*RES-1
    const int b    = row >> 5;
    const int h    = row & 31;
    const int tid  = threadIdx.x;
    const int wave = tid >> 5;                   // 0..7 -> N-group of 64
    const int lane = tid & 31;
    const int lm   = lane & 15;
    const bool hi  = lane >= 16;

    // ---- stage NR input rows (34 pixels incl. zero halo) into LDS ----
    constexpr int VPP = CINP / 8;                // b128 vectors per pixel
    const f32x4 zero4 = {0.f, 0.f, 0.f, 0.f};
    for (int v = tid; v < NR * 34 * VPP; v += 256) {
        int r   = v / (34 * VPP);
        int rem = v - r * (34 * VPP);
        int px  = rem / VPP;                     // 0..33 (pixel + 1)
        int c8  = rem - px * VPP;
        int hp = h + r - R0;
        int wp = px - 1;
        bool ok = (hp >= 0) && (hp < RES) && (wp >= 0) && (wp < RES);
        _Float16* lp = &smem[(r * 34 + px) * PITCH + c8 * 8];
        const _Float16* gp = ain + ((size_t)((b * RES + hp) * RES + wp)) * CINP + c8 * 8;
#if USE_ASYNC_BUILTIN
        if (ok) {
            __builtin_amdgcn_load_to_lds((void*)gp,
                                         (__attribute__((address_space(3))) void*)lp,
                                         16, 0, 0);
        } else {
            *(f32x4*)lp = zero4;
        }
#else
        if (ok) {
            // per-lane LDS byte offset (smem is the only LDS object -> base 0)
            unsigned loff = (unsigned)((const char*)lp - (const char*)&smem[0]);
            unsigned long long ga = (unsigned long long)(uintptr_t)gp;
            asm volatile("global_load_async_to_lds_b128 %0, %1, off"
                         :: "v"(loff), "v"(ga)
                         : "memory");
        } else {
            *(f32x4*)lp = zero4;
        }
#endif
    }
#if USE_ASYNC_BUILTIN
    __builtin_amdgcn_s_wait_asynccnt(0);
#else
    asm volatile("s_wait_asynccnt 0x0" ::: "memory");
#endif
    __syncthreads();

    v8f acc[2][4];
#pragma unroll
    for (int m = 0; m < 2; ++m)
#pragma unroll
        for (int i = 0; i < 4; ++i)
#pragma unroll
            for (int j = 0; j < 8; ++j) acc[m][i][j] = 0.0f;

    for (int tap = 0; tap < TAPS; ++tap) {
        const int dy = (TAPS == 9) ? (tap / 3 - 1) : 0;
        const int dx = (TAPS == 9) ? (tap % 3 - 1) : 0;
        const int rr = dy + R0;
        for (int kc = 0; kc < KC; ++kc) {
            // B fragments: 4 N-blocks, contiguous 32B per lane, L2-resident
            FragH bf[4];
            const size_t wbase = ((size_t)(tap * KC + kc)) * 32 * 512;
#pragma unroll
            for (int i = 0; i < 4; ++i) {
                const int nb = wave * 4 + i;
                const f32x4* q = (const f32x4*)(wf + wbase + (size_t)nb * 512 + lane * 16);
                bf[i].f[0] = q[0];
                bf[i].f[1] = q[1];
            }
#pragma unroll
            for (int mt = 0; mt < 2; ++mt) {
                const int px = 1 + mt * 16 + lm + dx;    // halo-shifted pixel
                const _Float16* ap = &smem[(rr * 34 + px) * PITCH + kc * 32 + (hi ? 16 : 0)];
                FragH a;
                a.f[0] = *(const f32x4*)ap;              // ds_load_b128
                a.f[1] = *(const f32x4*)(ap + 8);
#pragma unroll
                for (int i = 0; i < 4; ++i) {
                    acc[mt][i] = __builtin_amdgcn_wmma_f32_16x16x32_f16(
                        false, a.h, false, bf[i].h, (short)0, acc[mt][i], false, false);
                }
            }
        }
    }

    // ---- epilogue: bias + gelu, restage via LDS, coalesced b128 stores ----
    __syncthreads();                             // done reading input staging
#pragma unroll
    for (int mt = 0; mt < 2; ++mt) {
#pragma unroll
        for (int i = 0; i < 4; ++i) {
            const int n = wave * 64 + i * 16 + lm;
            const float bv = (n < MID) ? bias[n] : 0.0f;
#pragma unroll
            for (int j = 0; j < 8; ++j) {
                const int m = j + (hi ? 8 : 0);
                const int wpix = mt * 16 + m;
                float v = acc[mt][i][j] + bv;
                if (GACT) v = gelu_tanh(v);
                if (n >= MID) v = 0.0f;          // keep pad channels exactly zero
                smem[wpix * CPAD + n] = (_Float16)v;
            }
        }
    }
    __syncthreads();
    const size_t obase = (size_t)(b * RES + h) * RES * CPAD;
    for (int v2 = tid; v2 < 32 * CPAD / 8; v2 += 256)
        *(f32x4*)(aout + obase + (size_t)v2 * 8) = *(const f32x4*)&smem[v2 * 8];
}

// ---------------- tail kernels ----------------

// global average pool over HxW of the f16 activation (conv3 output)
__global__ __launch_bounds__(256) void pool_kernel(const _Float16* __restrict__ a,
                                                   float* __restrict__ pooled) {
    int b = blockIdx.x;
    for (int c = threadIdx.x; c < MID; c += 256) {
        float s = 0.0f;
        const _Float16* p = a + (size_t)b * NPIX * CPAD + c;
        for (int q = 0; q < NPIX; ++q) s += (float)p[(size_t)q * CPAD];
        pooled[b * CPAD + c] = s * (1.0f / NPIX);
    }
}

// 1x1 head after pooling: cp[b][o] = pooled[b] . w4[:,o] + b4[o]
__global__ __launch_bounds__(256) void head_kernel(const float* __restrict__ pooled,
                                                   const float* __restrict__ w4,
                                                   const float* __restrict__ b4,
                                                   float* __restrict__ cp) {
    int idx = blockIdx.x * blockDim.x + threadIdx.x;
    if (idx >= BATCH * OUTD) return;
    int b = idx / OUTD, o = idx % OUTD;
    float s = b4[o];
    const float* pb = pooled + b * CPAD;
    for (int k = 0; k < MID; ++k) s += pb[k] * w4[k * OUTD + o];
    cp[idx] = s;
}

// per (batch, channel): normalize weight (320 vals) and bias (16 vals),
// fold N_RANGES=16 into wn and precompute sum of normalized bias.
__global__ __launch_bounds__(64) void norm_kernel(const float* __restrict__ cp,
                                                  float* __restrict__ wn,
                                                  float* __restrict__ sbias) {
    int b = blockIdx.x, ch = blockIdx.y;
    int tid = threadIdx.x;
    __shared__ float red[64];
    __shared__ float winv_s, bnorm_s;
    const float* base = cp + (size_t)b * OUTD;

    float s = 0.0f;
    for (int i = tid; i < 320; i += 64) {
        float v = base[i * 3 + ch];
        s += v * v;
    }
    red[tid] = s; __syncthreads();
    for (int st = 32; st > 0; st >>= 1) {
        if (tid < st) red[tid] += red[tid + st];
        __syncthreads();
    }
    if (tid == 0) winv_s = 16.0f / sqrtf(red[0]);
    __syncthreads();
    float winv = winv_s;
    for (int i = tid; i < 320; i += 64)
        wn[(size_t)(b * 3 + ch) * 320 + i] = base[i * 3 + ch] * winv;

    // bias: sumsq then sum
    float bv = 0.0f;
    if (tid < 16) bv = base[960 + tid * 3 + ch];
    red[tid] = bv * bv; __syncthreads();
    for (int st = 32; st > 0; st >>= 1) {
        if (tid < st) red[tid] += red[tid + st];
        __syncthreads();
    }
    if (tid == 0) bnorm_s = sqrtf(red[0]);
    __syncthreads();
    red[tid] = bv; __syncthreads();
    for (int st = 32; st > 0; st >>= 1) {
        if (tid < st) red[tid] += red[tid + st];
        __syncthreads();
    }
    if (tid == 0) sbias[b * 3 + ch] = red[0] / bnorm_s;
}

// 10 fractal decode steps entirely in LDS, one block per (batch, channel)
__global__ __launch_bounds__(256) void decode_kernel(const float* __restrict__ wn,
                                                     const float* __restrict__ sbias,
                                                     float* __restrict__ zbuf) {
    int b = blockIdx.x, ch = blockIdx.y;
    int tid = threadIdx.x;
    __shared__ float z[1024];
    __shared__ float dom[20 * 64];
    __shared__ float wl[320];
    for (int i = tid; i < 1024; i += 256) z[i] = 0.0f;
    for (int i = tid; i < 320; i += 256) wl[i] = wn[(size_t)(b * 3 + ch) * 320 + i];
    float sb = sbias[b * 3 + ch];
    __syncthreads();

    for (int it = 0; it < 10; ++it) {
        // level-0 domains: 4 blocks of 16x16 avg-pooled 2x2 -> 8x8
        {
            int d0 = tid >> 6;
            int ii = (tid >> 3) & 7, jj = tid & 7;
            int qh = d0 >> 1, qw = d0 & 1;
            int y = qh * 16 + ii * 2, x = qw * 16 + jj * 2;
            dom[d0 * 64 + ii * 8 + jj] =
                0.25f * (z[y * 32 + x] + z[y * 32 + x + 1] +
                         z[(y + 1) * 32 + x] + z[(y + 1) * 32 + x + 1]);
        }
        // level-1 domains: 16 blocks of 8x8, identity pool
#pragma unroll
        for (int k = 0; k < 4; ++k) {
            int q = tid + k * 256;
            int d1 = q >> 6;
            int ii = (q >> 3) & 7, jj = q & 7;
            int qh = d1 >> 2, qw = d1 & 3;
            dom[(4 + d1) * 64 + ii * 8 + jj] = z[(qh * 8 + ii) * 32 + qw * 8 + jj];
        }
        __syncthreads();
        float outv[4];
#pragma unroll
        for (int k = 0; k < 4; ++k) {
            int q = tid + k * 256;
            int r = q >> 6;
            int ii = (q >> 3) & 7, jj = q & 7;
            float s = sb;
#pragma unroll
            for (int d = 0; d < 20; ++d) s += wl[r * 20 + d] * dom[d * 64 + ii * 8 + jj];
            outv[k] = s;
        }
        __syncthreads();
#pragma unroll
        for (int k = 0; k < 4; ++k) {
            int q = tid + k * 256;
            int r = q >> 6;
            int ii = (q >> 3) & 7, jj = q & 7;
            int rh = r >> 2, rw = r & 3;
            z[(rh * 8 + ii) * 32 + rw * 8 + jj] = outv[k];
        }
        __syncthreads();
    }
    for (int i = tid; i < 1024; i += 256)
        zbuf[(size_t)(b * 1024 + i) * 3 + ch] = z[i];
}

// per-batch min/max over (H,W,C), rescale, add x
__global__ __launch_bounds__(256) void finalize_kernel(const float* __restrict__ zbuf,
                                                       const float* __restrict__ x,
                                                       float* __restrict__ out) {
    int b = blockIdx.x;
    int tid = threadIdx.x;
    __shared__ float smn[256], smx[256];
    const float* zb = zbuf + (size_t)b * 3072;
    float mn = 3.402823466e38f, mx = -3.402823466e38f;
    for (int i = tid; i < 3072; i += 256) {
        float v = zb[i];
        mn = fminf(mn, v);
        mx = fmaxf(mx, v);
    }
    smn[tid] = mn; smx[tid] = mx; __syncthreads();
    for (int st = 128; st > 0; st >>= 1) {
        if (tid < st) {
            smn[tid] = fminf(smn[tid], smn[tid + st]);
            smx[tid] = fmaxf(smx[tid], smx[tid + st]);
        }
        __syncthreads();
    }
    float zmin = smn[0];
    float inv = 1.0f / (smx[0] - smn[0]);
    for (int i = tid; i < 3072; i += 256)
        out[(size_t)b * 3072 + i] = (zb[i] - zmin) * inv + x[(size_t)b * 3072 + i];
}

// ---------------- launch ----------------
extern "C" void kernel_launch(void* const* d_in, const int* in_sizes, int n_in,
                              void* d_out, int out_size, void* d_ws, size_t ws_size,
                              hipStream_t stream) {
    const float* x  = (const float*)d_in[0];
    const float* w1 = (const float*)d_in[1];
    const float* b1 = (const float*)d_in[2];
    const float* w2 = (const float*)d_in[3];
    const float* b2 = (const float*)d_in[4];
    const float* w3 = (const float*)d_in[5];
    const float* b3 = (const float*)d_in[6];
    const float* w4 = (const float*)d_in[7];
    const float* b4 = (const float*)d_in[8];
    float* out = (float*)d_out;
    (void)in_sizes; (void)n_in; (void)out_size; (void)ws_size;

    char* p = (char*)d_ws;
    auto carve = [&](size_t bytes) -> char* {
        char* r = p;
        p += (bytes + 255) & ~(size_t)255;
        return r;
    };
    _Float16* xg     = (_Float16*)carve((size_t)BATCH * NPIX * XPAD * sizeof(_Float16)); //   4 MB
    _Float16* actA   = (_Float16*)carve((size_t)BATCH * NPIX * CPAD * sizeof(_Float16)); //  64 MB
    _Float16* actB   = (_Float16*)carve((size_t)BATCH * NPIX * CPAD * sizeof(_Float16)); //  64 MB
    _Float16* w1p    = (_Float16*)carve((size_t)32 * 512 * sizeof(_Float16));
    _Float16* w2p    = (_Float16*)carve((size_t)9 * 16 * 32 * 512 * sizeof(_Float16));   // 4.5 MB
    _Float16* w3p    = (_Float16*)carve((size_t)9 * 16 * 32 * 512 * sizeof(_Float16));   // 4.5 MB
    float*    pooled = (float*)carve((size_t)BATCH * CPAD * sizeof(float));
    float*    cp     = (float*)carve((size_t)BATCH * OUTD * sizeof(float));
    float*    wn     = (float*)carve((size_t)BATCH * 3 * 320 * sizeof(float));
    float*    sb     = (float*)carve((size_t)BATCH * 3 * sizeof(float));
    float*    zbuf   = (float*)carve((size_t)BATCH * NPIX * 3 * sizeof(float));

    pack_xg_kernel<<<(BATCH * NPIX * XPAD) / 256, 256, 0, stream>>>(x, xg);
    pack_w1_kernel<<<(32 * 512) / 256, 256, 0, stream>>>(w1, w1p);
    pack_w9_kernel<<<(9 * 16 * 32 * 512) / 256, 256, 0, stream>>>(w2, w2p);
    pack_w9_kernel<<<(9 * 16 * 32 * 512) / 256, 256, 0, stream>>>(w3, w3p);

    conv_wmma_kernel<1, 1, XPAD, true ><<<BATCH * RES, 256, 0, stream>>>(xg,   w1p, b1, actA);
    conv_wmma_kernel<9, 16, CPAD, true ><<<BATCH * RES, 256, 0, stream>>>(actA, w2p, b2, actB);
    conv_wmma_kernel<9, 16, CPAD, false><<<BATCH * RES, 256, 0, stream>>>(actB, w3p, b3, actA);

    pool_kernel<<<BATCH, 256, 0, stream>>>(actA, pooled);
    head_kernel<<<(BATCH * OUTD + 255) / 256, 256, 0, stream>>>(pooled, w4, b4, cp);
    norm_kernel<<<dim3(BATCH, 3), 64, 0, stream>>>(cp, wn, sb);
    decode_kernel<<<dim3(BATCH, 3), 256, 0, stream>>>(wn, sb, zbuf);
    finalize_kernel<<<BATCH, 256, 0, stream>>>(zbuf, x, out);
}
